// SolarOpenAttention_55662776156253
// MI455X (gfx1250) — matrix operand order repaired and checked by reading, not compiled, not run on hardware
//
#include <hip/hip_runtime.h>
#include <hip/hip_bf16.h>

typedef __bf16 bf16;
typedef __attribute__((ext_vector_type(16))) __bf16 v16bf;
typedef __attribute__((ext_vector_type(8)))  __bf16 v8bf;
typedef __attribute__((ext_vector_type(8)))  float  v8f;

#define S_LEN 2048
#define HID   4096
#define NH    32
#define NKV   8
#define HD    128
#define QD    4096   /* NH*HD  */
#define KVD   1024   /* NKV*HD */
#define ATTN_SCALE 0.08838834764831845f  /* 1/sqrt(128), folded into Q at RoPE */

// K-offset within a 16x32 bf16 A-tile for vector element e, lane-half "half"
// (contiguous order, matching the B-fragment loads):
//   lanes 0-15 : elements 0..15 -> K=0..15
//   lanes 16-31: elements 0..15 -> K=16..31
__device__ __forceinline__ int a_kk(int e, int half) {
  return (half << 4) + e;
}
// Inverse: (row M in 0..15, K-offset kk in 0..31) -> (fragment lane, element)
__device__ __forceinline__ void a_pos(int Mrow, int kk, int& fragLane, int& e) {
  int half = (kk >> 4) & 1;
  e = kk & 15;
  fragLane = (half << 4) + Mrow;
}

__device__ __forceinline__ float rmax16(float v) {
  v = fmaxf(v, __shfl_xor(v, 8, 32));
  v = fmaxf(v, __shfl_xor(v, 4, 32));
  v = fmaxf(v, __shfl_xor(v, 2, 32));
  v = fmaxf(v, __shfl_xor(v, 1, 32));
  return v;
}
__device__ __forceinline__ float rsum16(float v) {
  v += __shfl_xor(v, 8, 32);
  v += __shfl_xor(v, 4, 32);
  v += __shfl_xor(v, 2, 32);
  v += __shfl_xor(v, 1, 32);
  return v;
}

// ---------------------------------------------------------------------------
// 1) f32 [M,K] -> bf16 packed into WMMA A-fragment tile order.
// ---------------------------------------------------------------------------
__global__ void pack_a_kernel(const float* __restrict__ X, bf16* __restrict__ Xa,
                              int Kcols) {
  int idx  = blockIdx.x * blockDim.x + threadIdx.x;
  int t    = idx >> 9;
  int rr   = idx & 511;
  int lane = rr >> 4, e = rr & 15;
  int Kt   = Kcols >> 5;
  int tm   = t / Kt, tk = t - tm * Kt;
  int row  = (tm << 4) + (lane & 15);
  int half = lane >> 4;
  int kk   = a_kk(e, half);
  Xa[idx] = (bf16)X[row * Kcols + (tk << 5) + kk];
}

// ---------------------------------------------------------------------------
// 2) f32 W[R,C] -> bf16 Wt[C,R] (transpose + convert) through LDS.
// ---------------------------------------------------------------------------
__global__ void transpose_cvt_kernel(const float* __restrict__ W,
                                     bf16* __restrict__ Wt, int R, int C) {
  __shared__ float tile[32][33];
  int tx = threadIdx.x, ty = threadIdx.y;
  int c0 = blockIdx.x * 32, r0 = blockIdx.y * 32;
#pragma unroll
  for (int i = 0; i < 4; ++i)
    tile[ty + i * 8][tx] = W[(size_t)(r0 + ty + i * 8) * C + c0 + tx];
  __syncthreads();
#pragma unroll
  for (int i = 0; i < 4; ++i)
    Wt[(size_t)(c0 + ty + i * 8) * R + r0 + tx] = (bf16)tile[tx][ty + i * 8];
}

// ---------------------------------------------------------------------------
// 3) GEMM: C[M,N] = A[M,K] x B[K,N].  A = packed bf16 fragments, B stored
//    transposed bf16 [N,K] so B-fragments are contiguous 32B loads.
//    OUTT=true writes C transposed ([N,M]) -- used for the V projection.
// ---------------------------------------------------------------------------
template <bool OUTF32, bool OUTT>
__global__ void gemm_bf16_kernel(const bf16* __restrict__ Ap,
                                 const bf16* __restrict__ Wt,
                                 void* __restrict__ Cout,
                                 int M, int N, int K) {
  int lane = threadIdx.x & 31;
  int wave = threadIdx.x >> 5;
  int wm = wave & 3, wn = wave >> 2;
  int bm = blockIdx.y, bn = blockIdx.x;
  int Kt = K >> 5;
  int half = lane >> 4;

  v8f acc00 = {}, acc01 = {}, acc10 = {}, acc11 = {};

  size_t tA0 = (size_t)(bm * 8 + wm * 2) * Kt;
  const bf16* bp0 =
      Wt + (size_t)(bn * 64 + wn * 32 + (lane & 15)) * K + (half << 4);
  const bf16* bp1 = bp0 + (size_t)16 * K;
  const bf16* ap0 = Ap + tA0 * 512 + lane * 16;
  const bf16* ap1 = ap0 + (size_t)Kt * 512;

#pragma unroll 2
  for (int ks = 0; ks < Kt; ++ks) {
    v16bf a0 = *(const v16bf*)(ap0 + (size_t)ks * 512);
    v16bf a1 = *(const v16bf*)(ap1 + (size_t)ks * 512);
    v16bf b0 = *(const v16bf*)(bp0 + (ks << 5));
    v16bf b1 = *(const v16bf*)(bp1 + (ks << 5));
    acc00 = __builtin_amdgcn_wmma_f32_16x16x32_bf16(false, a0, false, b0,
                                                    (short)0, acc00, false, false);
    acc01 = __builtin_amdgcn_wmma_f32_16x16x32_bf16(false, a0, false, b1,
                                                    (short)0, acc01, false, false);
    acc10 = __builtin_amdgcn_wmma_f32_16x16x32_bf16(false, a1, false, b0,
                                                    (short)0, acc10, false, false);
    acc11 = __builtin_amdgcn_wmma_f32_16x16x32_bf16(false, a1, false, b1,
                                                    (short)0, acc11, false, false);
  }

  int colBase = bn * 64 + wn * 32 + (lane & 15);
  int rowBase = bm * 128 + wm * 32 + (half << 3);
#pragma unroll
  for (int i = 0; i < 2; ++i) {
#pragma unroll
    for (int j = 0; j < 2; ++j) {
      v8f acc = (i == 0) ? (j == 0 ? acc00 : acc01) : (j == 0 ? acc10 : acc11);
#pragma unroll
      for (int r = 0; r < 8; ++r) {
        int row = rowBase + i * 16 + r;
        int col = colBase + j * 16;
        size_t o = OUTT ? (size_t)col * M + row : (size_t)row * N + col;
        if (OUTF32) ((float*)Cout)[o] = acc[r];
        else        ((bf16*)Cout)[o] = (bf16)acc[r];
      }
    }
  }
}

// ---------------------------------------------------------------------------
// 4) RoPE in place on bf16 [S, heads*128]; position = row index (arange).
//    outScale folds the attention 1/sqrt(HD) into Q (1.0 for K).
// ---------------------------------------------------------------------------
__global__ void rope_kernel(bf16* __restrict__ buf, int heads, float outScale) {
  int idx = blockIdx.x * blockDim.x + threadIdx.x;
  int d = idx & 63;
  int rest = idx >> 6;
  int hh = rest % heads;
  int q  = rest / heads;
  bf16* p = buf + (size_t)(q * heads + hh) * 128;
  float inv = __powf(10000.0f, -(float)d * (1.0f / 64.0f));
  float fr = (float)q * inv;
  float c = cosf(fr), s = sinf(fr);
  float x0 = (float)p[d], x1 = (float)p[d + 64];
  p[d]      = (bf16)((x0 * c - x1 * s) * outScale);
  p[d + 64] = (bf16)((x1 * c + x0 * s) * outScale);
}

// ---------------------------------------------------------------------------
// 5) Flash attention tile body. MASK=false for fully-causal-valid tiles
//    (kb+31 <= qbase): skips all per-element compare/select work.
// ---------------------------------------------------------------------------
template <bool MASK>
__device__ __forceinline__ void kv_tile(
    int kb, int qbase, int lane, int half, int l15,
    const bf16* __restrict__ kbase,   // Kb + l15*KVD + kvh*128 + half*16
    const bf16* __restrict__ vbase,   // Vt + (kvh*128+l15)*S_LEN + half*16
    bf16* ldsQ, bf16* ldsP,
    float (&m)[8], float (&l)[8], v8f (&o)[8]) {
  // ---- scores: two 16x16 C tiles over 32 keys ----
  v8f s0 = {}, s1 = {};
#pragma unroll
  for (int nt = 0; nt < 2; ++nt) {
    const bf16* kp = kbase + (size_t)(kb + nt * 16) * KVD;
    v8f sc = (nt == 0) ? s0 : s1;
#pragma unroll
    for (int ks = 0; ks < 4; ++ks) {
      v16bf aqf = *(const v16bf*)&ldsQ[ks * 512 + lane * 16];
      v16bf bk  = *(const v16bf*)(kp + (ks << 5));
      sc = __builtin_amdgcn_wmma_f32_16x16x32_bf16(false, aqf, false, bk,
                                                   (short)0, sc, false, false);
    }
    if (nt == 0) s0 = sc; else s1 = sc;
  }

  // ---- (mask) + online softmax; exp -> LDS P tile ----
  float corr[8];
#pragma unroll
  for (int r = 0; r < 8; ++r) {
    int Mrow = r + (half << 3);
    float v0 = s0[r];
    float v1 = s1[r];
    if (MASK) {
      int qrow = qbase + Mrow;
      if (kb + l15 > qrow)      v0 = -1e30f;
      if (kb + 16 + l15 > qrow) v1 = -1e30f;
    }
    float tm = rmax16(fmaxf(v0, v1));
    float mn = fmaxf(m[r], tm);
    corr[r] = __expf(m[r] - mn);
    m[r] = mn;
    float e0 = __expf(v0 - mn), e1 = __expf(v1 - mn);
    l[r] = l[r] * corr[r] + rsum16(e0 + e1);
    int fl, e;
    a_pos(Mrow, l15, fl, e);       ldsP[fl * 16 + e] = (bf16)e0;
    a_pos(Mrow, 16 + l15, fl, e);  ldsP[fl * 16 + e] = (bf16)e1;
  }
#pragma unroll
  for (int dt = 0; dt < 8; ++dt)
#pragma unroll
    for (int r = 0; r < 8; ++r) o[dt][r] *= corr[r];
  __syncthreads();

  // ---- PV: O[16x128] += P[16x32] x V[32x128], V frags straight from L2 ----
  v16bf ap = *(const v16bf*)&ldsP[lane * 16];
#pragma unroll
  for (int dt = 0; dt < 8; ++dt) {
    v16bf bv = *(const v16bf*)(vbase + (size_t)(dt << 4) * S_LEN + kb);
    o[dt] = __builtin_amdgcn_wmma_f32_16x16x32_bf16(false, ap, false, bv,
                                                    (short)0, o[dt], false, false);
  }
  __syncthreads();
}

// ---------------------------------------------------------------------------
// Flash attention: 1 wave per (16-query tile, head). Causal, GQA (kv=h/4).
// ---------------------------------------------------------------------------
__global__ void flash_attn_kernel(const bf16* __restrict__ Qb,
                                  const bf16* __restrict__ Kb,
                                  const bf16* __restrict__ Vt,
                                  bf16* __restrict__ Oa) {
  __shared__ __align__(32) bf16 ldsQ[4 * 512];   // 4 x (16x32) Q A-fragments
  __shared__ __align__(32) bf16 ldsP[512];       // 16x32 P tile, A-frag layout

  const int lane  = threadIdx.x;
  const int half  = lane >> 4;
  const int l15   = lane & 15;
  const int qbase = blockIdx.x << 4;
  const int h     = blockIdx.y;
  const int kvh   = h >> 2;

  // --- Q A-fragments into LDS (one-time gather) ---
  {
    const bf16* qrow = Qb + (size_t)(qbase + l15) * QD + (h << 7);
#pragma unroll
    for (int ks = 0; ks < 4; ++ks)
#pragma unroll
      for (int e = 0; e < 16; ++e)
        ldsQ[ks * 512 + lane * 16 + e] = qrow[(ks << 5) + a_kk(e, half)];
  }
  __syncthreads();

  float m[8], l[8];
  v8f o[8];
#pragma unroll
  for (int r = 0; r < 8; ++r) { m[r] = -1e30f; l[r] = 0.0f; }
#pragma unroll
  for (int dt = 0; dt < 8; ++dt) o[dt] = (v8f){};

  const bf16* kbase = Kb + (size_t)l15 * KVD + (kvh << 7) + (half << 4);
  const bf16* vbase = Vt + (size_t)((kvh << 7) + l15) * S_LEN + (half << 4);

  const int nkt   = (qbase + 16 + 31) >> 5;  // total key tiles (keys <= qbase+15)
  const int nfull = qbase >> 5;              // tiles with kb+31 <= qbase: no mask
  int kt = 0;
  for (; kt < nfull; ++kt)
    kv_tile<false>(kt << 5, qbase, lane, half, l15, kbase, vbase, ldsQ, ldsP, m, l, o);
  for (; kt < nkt; ++kt)
    kv_tile<true>(kt << 5, qbase, lane, half, l15, kbase, vbase, ldsQ, ldsP, m, l, o);

  // ---- normalize + write in packed-A layout (rows = q, cols = h*128+d) ----
#pragma unroll
  for (int r = 0; r < 8; ++r) {
    int Mrow = r + (half << 3);
    int q = qbase + Mrow;
    float inv_l = 1.0f / l[r];
#pragma unroll
    for (int dt = 0; dt < 8; ++dt) {
      int col = (h << 7) + (dt << 4) + l15;
      float val = o[dt][r] * inv_l;
      int tile = (q >> 4) * (QD >> 5) + (col >> 5);
      int fl, e;
      a_pos(Mrow, col & 31, fl, e);
      Oa[(size_t)tile * 512 + fl * 16 + e] = (bf16)val;
    }
  }
}

// ---------------------------------------------------------------------------
// Host orchestration
// ---------------------------------------------------------------------------
extern "C" void kernel_launch(void* const* d_in, const int* in_sizes, int n_in,
                              void* d_out, int out_size, void* d_ws, size_t ws_size,
                              hipStream_t stream) {
  const float* X  = (const float*)d_in[0];
  // d_in[1] = position_ids (== arange(S)); positions derived from row index.
  const float* wq = (const float*)d_in[2];
  const float* wk = (const float*)d_in[3];
  const float* wv = (const float*)d_in[4];
  const float* wo = (const float*)d_in[5];

  char* ws = (char*)d_ws;
  size_t off = 0;
  auto alloc = [&](size_t bytes) -> void* {
    void* p = ws + off;
    off += (bytes + 255) & ~(size_t)255;
    return p;
  };
  bf16* Xa  = (bf16*)alloc((size_t)S_LEN * HID * 2);  // packed-A hidden states
  bf16* Wqt = (bf16*)alloc((size_t)QD  * HID * 2);    // [N,K] bf16
  bf16* Wkt = (bf16*)alloc((size_t)KVD * HID * 2);
  bf16* Wvt = (bf16*)alloc((size_t)KVD * HID * 2);
  bf16* Wot = (bf16*)alloc((size_t)HID * QD  * 2);
  bf16* Qb  = (bf16*)alloc((size_t)S_LEN * QD  * 2);  // [S, H*HD]
  bf16* Kb  = (bf16*)alloc((size_t)S_LEN * KVD * 2);  // [S, HKV*HD]
  bf16* Vt  = (bf16*)alloc((size_t)KVD * S_LEN * 2);  // [HKV*HD, S] transposed
  bf16* Oa  = (bf16*)alloc((size_t)S_LEN * QD  * 2);  // packed-A attn output
  (void)ws_size; (void)in_sizes; (void)n_in; (void)out_size;

  // 1) pack hidden states
  pack_a_kernel<<<(S_LEN * HID) / 256, 256, 0, stream>>>(X, Xa, HID);

  // 2) transpose+convert weights: W[K,N] f32 -> Wt[N,K] bf16
  transpose_cvt_kernel<<<dim3(QD / 32,  HID / 32), dim3(32, 8), 0, stream>>>(wq, Wqt, HID, QD);
  transpose_cvt_kernel<<<dim3(KVD / 32, HID / 32), dim3(32, 8), 0, stream>>>(wk, Wkt, HID, KVD);
  transpose_cvt_kernel<<<dim3(KVD / 32, HID / 32), dim3(32, 8), 0, stream>>>(wv, Wvt, HID, KVD);
  transpose_cvt_kernel<<<dim3(HID / 32, QD / 32),  dim3(32, 8), 0, stream>>>(wo, Wot, QD, HID);

  // 3) QKV projections (V written transposed: [KVD, S])
  gemm_bf16_kernel<false, false><<<dim3(QD / 64,  S_LEN / 128), 256, 0, stream>>>(Xa, Wqt, Qb, S_LEN, QD,  HID);
  gemm_bf16_kernel<false, false><<<dim3(KVD / 64, S_LEN / 128), 256, 0, stream>>>(Xa, Wkt, Kb, S_LEN, KVD, HID);
  gemm_bf16_kernel<false, true ><<<dim3(KVD / 64, S_LEN / 128), 256, 0, stream>>>(Xa, Wvt, Vt, S_LEN, KVD, HID);

  // 4) RoPE on Q (with 1/sqrt(HD) folded in) and K
  rope_kernel<<<(S_LEN * NH  * 64) / 256, 256, 0, stream>>>(Qb, NH, ATTN_SCALE);
  rope_kernel<<<(S_LEN * NKV * 64) / 256, 256, 0, stream>>>(Kb, NKV, 1.0f);

  // 5) causal GQA flash attention -> packed-A output
  flash_attn_kernel<<<dim3(S_LEN / 16, NH), 32, 0, stream>>>(Qb, Kb, Vt, Oa);

  // 6) output projection -> f32
  gemm_bf16_kernel<true, false><<<dim3(HID / 64, S_LEN / 128), 256, 0, stream>>>(Oa, Wot, d_out, S_LEN, HID, QD);
}
